// AttentionLayer_1623497637890
// MI455X (gfx1250) — matrix operand order repaired and checked by reading, not compiled, not run on hardware
//
#include <hip/hip_runtime.h>
#include <hip/hip_bf16.h>
#include <cstdint>
#include <cstddef>

// ---------------------------------------------------------------------------
// AttentionLayer for MI455X (gfx1250): bf16 WMMA everywhere, f32 accumulate.
//   B=4, N=4096, D=1024. Compute-bound (~0.38 TFLOP vs ~140MB HBM traffic).
//   q/k/vT in bf16 total 96MB -> resident in 192MB L2 during flash loop.
//   K tiles staged into LDS by the Tensor Data Mover (double-buffered,
//   TENSORcnt-synchronized) so the 4 waves of a workgroup share one copy.
// ---------------------------------------------------------------------------

typedef __attribute__((ext_vector_type(16))) __bf16 v16bf;
typedef __attribute__((ext_vector_type(8)))  __bf16 v8bf;
typedef __attribute__((ext_vector_type(8)))  float  v8f;
typedef __attribute__((ext_vector_type(4)))  float  v4f;

// TDM descriptor groups (clang-23 / therock-10.0: 6-arg builtin signature)
typedef __attribute__((ext_vector_type(4))) unsigned tdm_g0_t;
typedef __attribute__((ext_vector_type(8))) int      tdm_g1_t;
typedef __attribute__((ext_vector_type(4))) int      tdm_g2_t;
typedef __attribute__((ext_vector_type(8))) int      tdm_g4_t;

static constexpr int BATCH = 4;
static constexpr int SEQ   = 4096;
static constexpr int DIM   = 1024;
static constexpr int MTOT  = BATCH * SEQ;          // 16384 rows total

// ---- bf16 <-> f32 helpers (RNE), no reliance on __bf16 arithmetic ----------
__device__ __forceinline__ __bf16 f2bf(float f) {
  unsigned u = __builtin_bit_cast(unsigned, f);
  u += 0x7FFFu + ((u >> 16) & 1u);
  unsigned short h = (unsigned short)(u >> 16);
  return __builtin_bit_cast(__bf16, h);
}
__device__ __forceinline__ float bf2f(__bf16 b) {
  unsigned short h = __builtin_bit_cast(unsigned short, b);
  unsigned u = ((unsigned)h) << 16;
  return __builtin_bit_cast(float, u);
}

// ---- WMMA wrapper ----------------------------------------------------------
__device__ __forceinline__ v8f wmma_bf16(v16bf a, v16bf b, v8f c) {
  // D = A(16x32 bf16) * B(32x16 bf16) + C(16x16 f32)
  return __builtin_amdgcn_wmma_f32_16x16x32_bf16(
      /*neg_a=*/false, a, /*neg_b=*/false, b,
      /*c_mod=*/(short)0, c, /*reuse_a=*/false, /*reuse_b=*/false);
}

// ---- Fragment loaders matching CDNA5 ISA VGPR layouts ----------------------
// A 16x32 bf16: lanes 0-15 = rows, VGPR0-3 hold K {kb..kb+7}, VGPR4-7 hold
// K {kb+16..kb+23}; lanes 16-31 same rows with K {+8} offset.
__device__ __forceinline__ v16bf load_abf(const __bf16* base, int row0, int ld,
                                          int kb, int lane) {
  const int l = lane & 15, half = lane >> 4;
  const __bf16* p = base + (size_t)(row0 + l) * (size_t)ld + kb + half * 8;
  v8bf lo = *(const v8bf*)p;          // K kb..kb+7   (or +8..+15 hi half)
  v8bf hi = *(const v8bf*)(p + 16);   // K kb+16..+23 (or +24..+31 hi half)
  v16bf r;
#pragma unroll
  for (int i = 0; i < 8; ++i) { r[i] = lo[i]; r[i + 8] = hi[i]; }
  return r;
}

// B 32x16 bf16: lane n<16 = column col0+n, VGPR0-3 hold K {kb..kb+7}, VGPR4-7
// hold K {kb+16..kb+23}; lanes 16-31 same columns with K {+8} offset.
// Requires source stored as [column][K] row-major (stride ld).
__device__ __forceinline__ v16bf load_bbf(const __bf16* base, int col0, int ld,
                                          int kb, int lane) {
  const int l = lane & 15, half = lane >> 4;
  const __bf16* p = base + (size_t)(col0 + l) * (size_t)ld + kb + half * 8;
  v8bf lo = *(const v8bf*)p;
  v8bf hi = *(const v8bf*)(p + 16);
  v16bf r;
#pragma unroll
  for (int i = 0; i < 8; ++i) { r[i] = lo[i]; r[i + 8] = hi[i]; }
  return r;
}

// ---- TDM: DMA a 64-key x STG_D-dim bf16 tile of kh into LDS ----------------
// 2-D descriptor: tensor = [MTOT][DIM] bf16 row-major; tile = STG_D x 64.
// lds layout produced: [64 keys][STG_D dims] contiguous (x fastest).
static constexpr int STG_D = 128;                      // dims per stage
static constexpr int STG_BYTES = 64 * STG_D * 2;       // 16 KB per buffer

__device__ __forceinline__ void tdm_load_k_tile(const __bf16* gaddr,
                                                unsigned lds_off) {
  const unsigned long long ga = (unsigned long long)(uintptr_t)gaddr;
  tdm_g0_t g0;
  g0[0] = 1u;                                     // count=1, user descriptor
  g0[1] = lds_off;                                // lds_addr (bytes)
  g0[2] = (unsigned)(ga & 0xFFFFFFFFu);           // global_addr[31:0]
  g0[3] = (unsigned)((ga >> 32) & 0x01FFFFFFu)    // global_addr[56:32]
          | (2u << 30);                           // type = 2 ("image")
  tdm_g1_t g1;
  g1[0] = (int)(1u << 16);                        // data_size=1 -> 2 bytes
  g1[1] = (int)(((unsigned)DIM & 0xFFFFu) << 16); // tensor_dim0 lo16
  g1[2] = (int)((((unsigned)DIM >> 16) & 0xFFFFu)
          | (((unsigned)MTOT & 0xFFFFu) << 16));  // dim0 hi16 | dim1 lo16
  g1[3] = (int)((((unsigned)MTOT >> 16) & 0xFFFFu)
          | ((unsigned)STG_D << 16));             // dim1 hi16 | tile_dim0
  g1[4] = 64;                                     // tile_dim1=64, tile_dim2=0
  g1[5] = DIM;                                    // tensor_dim0_stride lo32
  g1[6] = 0;                                      // stride0 hi16 | stride1 lo
  g1[7] = 0;
  tdm_g2_t z4 = {0, 0, 0, 0};                     // 2-D: groups 2/3 unused
  tdm_g4_t z8 = {0, 0, 0, 0, 0, 0, 0, 0};         // extra group (unused)
  __builtin_amdgcn_tensor_load_to_lds(g0, g1, z4, z4, z8, 0);
}

// ---------------------------------------------------------------------------
// Kernel 1: elementwise f32 -> bf16 cast (8 elems / thread / iter)
// ---------------------------------------------------------------------------
__global__ void cast_kernel(const float* __restrict__ src,
                            __bf16* __restrict__ dst, int n8) {
  int i = blockIdx.x * blockDim.x + threadIdx.x;
  int stride = gridDim.x * blockDim.x;
  for (; i < n8; i += stride) {
    v8f xv = *((const v8f*)src + i);
    v8bf o;
#pragma unroll
    for (int k = 0; k < 8; ++k) o[k] = f2bf(xv[k]);
    *((v8bf*)dst + i) = o;
  }
}

// ---------------------------------------------------------------------------
// Kernel 2: q/k/v projections.  z = 0:q, 1:k, 2:v(transposed output).
// Workgroup = 4 waves; wave tile = 32(M) x 64(N); K-loop step 32 -> 8 WMMA.
// ---------------------------------------------------------------------------
__global__ __launch_bounds__(128) void qkv_kernel(
    const __bf16* __restrict__ xh,
    const __bf16* __restrict__ wqh, const __bf16* __restrict__ wkh,
    const __bf16* __restrict__ wvh,
    const float* __restrict__ bq, const float* __restrict__ bk,
    const float* __restrict__ bv,
    __bf16* __restrict__ qh, __bf16* __restrict__ kh,
    __bf16* __restrict__ vT) {
  const int lane = threadIdx.x & 31;
  const int wave = threadIdx.x >> 5;
  const int l = lane & 15, half = lane >> 4;
  const int z = blockIdx.z;
  const __bf16* W = (z == 0) ? wqh : (z == 1) ? wkh : wvh;
  const float* bias = (z == 0) ? bq : (z == 1) ? bk : bv;
  const int m0 = blockIdx.x * 128 + wave * 32;   // two 16-row M tiles
  const int e0 = blockIdx.y * 64;                // four 16-col N tiles

  v8f acc[2][4];
#pragma unroll
  for (int mt = 0; mt < 2; ++mt)
#pragma unroll
    for (int nt = 0; nt < 4; ++nt)
#pragma unroll
      for (int r = 0; r < 8; ++r) acc[mt][nt][r] = 0.f;

  for (int kb = 0; kb < DIM; kb += 32) {
    v16bf a0 = load_abf(xh, m0, DIM, kb, lane);
    v16bf a1 = load_abf(xh, m0 + 16, DIM, kb, lane);
#pragma unroll
    for (int nt = 0; nt < 4; ++nt) {
      // W[e][d] row-major is exactly [column][K] for x @ W^T
      v16bf bfrag = load_bbf(W, e0 + nt * 16, DIM, kb, lane);
      acc[0][nt] = wmma_bf16(a0, bfrag, acc[0][nt]);
      acc[1][nt] = wmma_bf16(a1, bfrag, acc[1][nt]);
    }
  }

#pragma unroll
  for (int mt = 0; mt < 2; ++mt)
#pragma unroll
    for (int nt = 0; nt < 4; ++nt) {
      const int col = e0 + nt * 16 + l;
      const float bval = bias[col];
#pragma unroll
      for (int r = 0; r < 8; ++r) {
        const int row = m0 + mt * 16 + r + 8 * half;
        const __bf16 h = f2bf(acc[mt][nt][r] + bval);
        if (z == 0)      qh[(size_t)row * DIM + col] = h;
        else if (z == 1) kh[(size_t)row * DIM + col] = h;
        else             vT[(size_t)col * MTOT + row] = h;   // transposed V
      }
    }
}

// ---------------------------------------------------------------------------
// Kernel 3: flash attention + residual.
//   Workgroup: 64 query rows, 4 waves (16 rows each for S; 256-dim output
//   chunk each for rescale/PV). O accumulator 64x1024 f32 lives in LDS.
//   K tiles (64 keys x 128 dims) DMA'd into LDS by the TDM, double-buffered.
// ---------------------------------------------------------------------------
static constexpr int SMEM_O   = 64 * 1024 * 4;               // 262144
static constexpr int SMEM_P   = 64 * 64 * 2;                 // 8192
static constexpr int SMEM_ST  = SMEM_O + SMEM_P;             // stats offset
static constexpr int SMEM_KS  = SMEM_ST + 2 * 64 * 4;        // 270848 (16B al)
static constexpr int SMEM_BYTES = SMEM_KS + 2 * STG_BYTES;   // 303616 < 320KB

__global__ __launch_bounds__(128) void attn_kernel(
    const __bf16* __restrict__ qh, const __bf16* __restrict__ kh,
    const __bf16* __restrict__ vT, const float* __restrict__ x,
    float* __restrict__ out) {
  extern __shared__ char smem[];
  float*  Os     = (float*)smem;                       // [64][1024] f32
  __bf16* Ps     = (__bf16*)(smem + SMEM_O);           // [64][64] bf16
  float*  alphas = (float*)(smem + SMEM_ST);           // [64]
  float*  ls     = alphas + 64;                        // [64]

  const int lane = threadIdx.x & 31, wave = threadIdx.x >> 5;
  const int l = lane & 15, half = lane >> 4;
  const int b  = blockIdx.y;
  const int q0 = blockIdx.x * 64;
  const int rowbase = b * SEQ;
  const int qrow0 = rowbase + q0 + wave * 16;
  const int dchunk = wave * 256;
  constexpr int NSTG = DIM / STG_D;   // 8 stages per key block

  // zero the O accumulator
  for (int i = threadIdx.x; i < 64 * 1024 / 4; i += 128) {
    v4f z = {0.f, 0.f, 0.f, 0.f};
    ((v4f*)Os)[i] = z;
  }
  __syncthreads();

  float m_r[8], l_r[8];
#pragma unroll
  for (int r = 0; r < 8; ++r) { m_r[r] = -__builtin_inff(); l_r[r] = 0.f; }

  for (int j = 0; j < SEQ; j += 64) {
    const int krow0 = rowbase + j;

    // ---- S = (Q K^T) : TDM stages K tiles into LDS, double-buffered --------
    if (wave == 0) {
      tdm_load_k_tile(kh + (size_t)krow0 * DIM + 0 * STG_D, SMEM_KS);
      tdm_load_k_tile(kh + (size_t)krow0 * DIM + 1 * STG_D,
                      SMEM_KS + STG_BYTES);
    }

    v8f s[4];
#pragma unroll
    for (int nt = 0; nt < 4; ++nt)
#pragma unroll
      for (int r = 0; r < 8; ++r) s[nt][r] = 0.f;

    for (int st = 0; st < NSTG; ++st) {
      if (wave == 0) {
        // TDM completes in-order per wave: <=1 outstanding => stage st done
        if (st < NSTG - 1) __builtin_amdgcn_s_wait_tensorcnt(1);
        else               __builtin_amdgcn_s_wait_tensorcnt(0);
      }
      __syncthreads();                       // staged tile visible to all

      // staged tile: [64 keys][STG_D dims]  (computed inline: no pointer
      // array of addrspacecasts -> avoids static-initializer lowering)
      const __bf16* kst =
          (const __bf16*)(smem + SMEM_KS + (st & 1) * STG_BYTES);
      const int kb0 = st * STG_D;
#pragma unroll
      for (int kk = 0; kk < STG_D; kk += 32) {
        v16bf a = load_abf(qh, qrow0, DIM, kb0 + kk, lane);
#pragma unroll
        for (int nt = 0; nt < 4; ++nt) {
          v16bf bb = load_bbf(kst, nt * 16, STG_D, kk, lane);
          s[nt] = wmma_bf16(a, bb, s[nt]);
        }
      }
      __syncthreads();                       // everyone done with buffer
      if (wave == 0 && st + 2 < NSTG)        // refill the buffer just freed
        tdm_load_k_tile(kh + (size_t)krow0 * DIM + (st + 2) * STG_D,
                        SMEM_KS + (st & 1) * STG_BYTES);
    }

    // ---- online softmax (scale = 1/sqrt(1024) = 0.03125 exactly) -----------
    float pv[4][8], asc[8];
#pragma unroll
    for (int r = 0; r < 8; ++r) {
      float mx = -__builtin_inff();
#pragma unroll
      for (int nt = 0; nt < 4; ++nt) {
        float v = s[nt][r] * 0.03125f;
        pv[nt][r] = v;
        mx = fmaxf(mx, v);
      }
      for (int off = 8; off >= 1; off >>= 1)
        mx = fmaxf(mx, __shfl_xor(mx, off, 16));   // reduce within 16-lane half
      const float mnew = fmaxf(m_r[r], mx);
      asc[r] = __expf(m_r[r] - mnew);
      float sum = 0.f;
#pragma unroll
      for (int nt = 0; nt < 4; ++nt) {
        float e = __expf(pv[nt][r] - mnew);
        pv[nt][r] = e;
        sum += e;
      }
      for (int off = 8; off >= 1; off >>= 1) sum += __shfl_xor(sum, off, 16);
      l_r[r] = l_r[r] * asc[r] + sum;
      m_r[r] = mnew;
    }

    // write P (bf16) and alpha for this wave's rows
#pragma unroll
    for (int nt = 0; nt < 4; ++nt)
#pragma unroll
      for (int r = 0; r < 8; ++r)
        Ps[(wave * 16 + r + 8 * half) * 64 + nt * 16 + l] = f2bf(pv[nt][r]);
    if (l == 0)
#pragma unroll
      for (int r = 0; r < 8; ++r) alphas[wave * 16 + 8 * half + r] = asc[r];
    __syncthreads();

    // ---- rescale this wave's O chunk by per-row alpha ----------------------
    for (int row = 0; row < 64; ++row) {
      const float av = alphas[row];
      float* po = Os + row * 1024 + dchunk;
#pragma unroll
      for (int t = 0; t < 2; ++t) {
        v4f* p = (v4f*)po + lane + t * 32;
        v4f v = *p;
#pragma unroll
        for (int i = 0; i < 4; ++i) v[i] *= av;
        *p = v;
      }
    }

    // ---- O += P @ V : wave covers all 64 rows x its 256 dims ---------------
    v16bf pa[4][2];
#pragma unroll
    for (int mt = 0; mt < 4; ++mt) {
      pa[mt][0] = load_abf(Ps, mt * 16, 64, 0, lane);
      pa[mt][1] = load_abf(Ps, mt * 16, 64, 32, lane);
    }
    for (int nt = 0; nt < 16; ++nt) {
      const int d0 = dchunk + nt * 16;
      v16bf vb0 = load_bbf(vT, d0, MTOT, krow0, lane);       // keys 0..31
      v16bf vb1 = load_bbf(vT, d0, MTOT, krow0 + 32, lane);  // keys 32..63
#pragma unroll
      for (int mt = 0; mt < 4; ++mt) {
        float* pc = Os + (mt * 16 + half * 8) * 1024 + d0 + l;
        v8f c;
#pragma unroll
        for (int r = 0; r < 8; ++r) c[r] = pc[r * 1024];
        c = wmma_bf16(pa[mt][0], vb0, c);
        c = wmma_bf16(pa[mt][1], vb1, c);
#pragma unroll
        for (int r = 0; r < 8; ++r) pc[r * 1024] = c[r];
      }
    }
    __syncthreads();   // protect Ps/alphas/stages for next key block
  }

  // ---- epilogue: out = O / l + x ------------------------------------------
  if (l == 0)
#pragma unroll
    for (int r = 0; r < 8; ++r) ls[wave * 16 + 8 * half + r] = l_r[r];
  __syncthreads();

  for (int row = 0; row < 64; ++row) {
    const float inv = 1.0f / ls[row];
    const size_t g = (size_t)(rowbase + q0 + row) * DIM + dchunk;
#pragma unroll
    for (int t = 0; t < 2; ++t) {
      const int off = t * 128 + lane * 4;
      v4f o = *(v4f*)(Os + row * 1024 + dchunk + off);
      v4f xv = *(const v4f*)(x + g + off);
      v4f rr;
#pragma unroll
      for (int i = 0; i < 4; ++i) rr[i] = o[i] * inv + xv[i];
      *(v4f*)(out + g + off) = rr;
    }
  }
}

// ---------------------------------------------------------------------------
extern "C" void kernel_launch(void* const* d_in, const int* in_sizes, int n_in,
                              void* d_out, int out_size, void* d_ws,
                              size_t ws_size, hipStream_t stream) {
  (void)in_sizes; (void)n_in; (void)out_size; (void)ws_size;
  const float* x  = (const float*)d_in[0];
  const float* Wq = (const float*)d_in[1];
  const float* bq = (const float*)d_in[2];
  const float* Wk = (const float*)d_in[3];
  const float* bk = (const float*)d_in[4];
  const float* Wv = (const float*)d_in[5];
  const float* bv = (const float*)d_in[6];
  float* out = (float*)d_out;

  // workspace layout (bf16 staging, ~140MB)
  char* ws = (char*)d_ws;
  __bf16* xh  = (__bf16*)ws;  ws += (size_t)MTOT * DIM * 2;
  __bf16* wqh = (__bf16*)ws;  ws += (size_t)DIM * DIM * 2;
  __bf16* wkh = (__bf16*)ws;  ws += (size_t)DIM * DIM * 2;
  __bf16* wvh = (__bf16*)ws;  ws += (size_t)DIM * DIM * 2;
  __bf16* qh  = (__bf16*)ws;  ws += (size_t)MTOT * DIM * 2;
  __bf16* kh  = (__bf16*)ws;  ws += (size_t)MTOT * DIM * 2;
  __bf16* vT  = (__bf16*)ws;  ws += (size_t)MTOT * DIM * 2;

  // 1) casts
  {
    int n8 = MTOT * DIM / 8;
    cast_kernel<<<2048, 256, 0, stream>>>(x, xh, n8);
    int w8 = DIM * DIM / 8;
    cast_kernel<<<512, 256, 0, stream>>>(Wq, wqh, w8);
    cast_kernel<<<512, 256, 0, stream>>>(Wk, wkh, w8);
    cast_kernel<<<512, 256, 0, stream>>>(Wv, wvh, w8);
  }

  // 2) q/k/v projections
  {
    dim3 grid(MTOT / 128, DIM / 64, 3);
    qkv_kernel<<<grid, 128, 0, stream>>>(xh, wqh, wkh, wvh, bq, bk, bv,
                                         qh, kh, vT);
  }

  // 3) flash attention + residual (~297KB dynamic LDS; CDNA5: 320KB/WG)
  {
    (void)hipFuncSetAttribute((const void*)attn_kernel,
                              hipFuncAttributeMaxDynamicSharedMemorySize,
                              SMEM_BYTES);
    dim3 grid(SEQ / 64, BATCH);
    attn_kernel<<<grid, 128, SMEM_BYTES, stream>>>(qh, kh, vT, x, out);
  }
}
